// SelfAttention_343597384103
// MI455X (gfx1250) — matrix operand order repaired and checked by reading, not compiled, not run on hardware
//
#include <hip/hip_runtime.h>
#include <stdint.h>

#define S_LEN 4096
#define D_DIM 1024
#define H_CNT 16
#define DH    64
#define F_DIM 4096
#define LN_EPS 1e-5f
#define SCORE_SCALE 0.125f   // DH^-0.5

typedef unsigned short us;
typedef __attribute__((ext_vector_type(16))) __bf16 v16bf;
typedef __attribute__((ext_vector_type(8)))  float  v8f;
typedef __attribute__((address_space(3))) us lds_us;

union ABFrag { v16bf v; uint4 q[2]; };
union U8     { uint4 q; us s[8]; };

__device__ inline us f2bf(float f) {
  unsigned u = __float_as_uint(f);
  u += 0x7fffu + ((u >> 16) & 1u);          // round-to-nearest-even
  return (us)(u >> 16);
}

__device__ inline v8f wmma_bf16(v16bf a, v16bf b, v8f c) {
  // D = A(16x32 bf16) * B(32x16 bf16) + C(16x16 f32)
  return __builtin_amdgcn_wmma_f32_16x16x32_bf16(
      false, a, false, b, (short)0, c, false, false);
}

__device__ inline v8f vzero8() {
  v8f z;
#pragma unroll
  for (int i = 0; i < 8; i++) z[i] = 0.0f;
  return z;
}

// Async DMA: global (16B) -> LDS, tracked by ASYNCcnt (CDNA5 TDM-lite path).
__device__ inline void async_ldg_b128(const us* gsrc, us* lds_dst) {
  lds_us* p3 = (lds_us*)lds_dst;                       // generic -> AS(3)
  unsigned ldsa = (unsigned)(unsigned long long)p3;    // 32-bit LDS address
  unsigned long long ga = (unsigned long long)gsrc;
  asm volatile("global_load_async_to_lds_b128 %0, %1, off"
               :: "v"(ldsa), "v"(ga) : "memory");
}
__device__ inline void wait_asynccnt0() {
  asm volatile("s_wait_asynccnt 0" ::: "memory");
}

// ---------------------------------------------------------------------------
// fp32 -> bf16 conversion (weights / staging)
// ---------------------------------------------------------------------------
__global__ void f32_to_bf16_kernel(const float* __restrict__ in,
                                   us* __restrict__ out, long n) {
  long i = (long)blockIdx.x * blockDim.x + threadIdx.x;
  long stride = (long)gridDim.x * blockDim.x;
  for (; i < n; i += stride) out[i] = f2bf(in[i]);
}

// ---------------------------------------------------------------------------
// LayerNorm: one row (D=1024) per 256-thread block, bf16 output
// ---------------------------------------------------------------------------
__global__ __launch_bounds__(256)
void layernorm_bf16_kernel(const float* __restrict__ x,
                           const float* __restrict__ g,
                           const float* __restrict__ b,
                           us* __restrict__ out) {
  __shared__ float s1[256], s2[256];
  const int row = blockIdx.x;
  const int tid = threadIdx.x;
  const float* xr = x + (long)row * D_DIM;
  float v[4]; float sum = 0.f, sq = 0.f;
#pragma unroll
  for (int i = 0; i < 4; i++) {
    v[i] = xr[tid + i * 256];
    sum += v[i]; sq += v[i] * v[i];
  }
  s1[tid] = sum; s2[tid] = sq;
  __syncthreads();
  for (int st = 128; st > 0; st >>= 1) {
    if (tid < st) { s1[tid] += s1[tid + st]; s2[tid] += s2[tid + st]; }
    __syncthreads();
  }
  const float mean = s1[0] * (1.0f / D_DIM);
  const float var  = s2[0] * (1.0f / D_DIM) - mean * mean;
  const float rs   = rsqrtf(var + LN_EPS);
#pragma unroll
  for (int i = 0; i < 4; i++) {
    int c = tid + i * 256;
    out[(long)row * D_DIM + c] = f2bf((v[i] - mean) * rs * g[c] + b[c]);
  }
}

// ---------------------------------------------------------------------------
// Tiled bf16 WMMA GEMM:  C[M,N] = A[M,K] @ B[K,N] + bias (+gelu) (+residual)
// Block: 256 thr = 8 waves, tile 128x128, wave grid 4(M) x 2(N) (32x64/wave),
// K-step 32, double-buffered LDS; A staged by async-DMA, B register-transposed.
// ---------------------------------------------------------------------------
__global__ __launch_bounds__(256)
void gemm_bf16_kernel(const us* __restrict__ A, const us* __restrict__ B,
                      const float* __restrict__ bias,
                      const float* __restrict__ resid,
                      float* __restrict__ outF, us* __restrict__ outB,
                      int M, int N, int K, int gelu) {
  __shared__ us As[2][128 * 32];  // [row][k] row-major
  __shared__ us Bs[2][128 * 32];  // transposed: [n][k]
  const int tid  = threadIdx.x;
  const int lane = tid & 31;
  const int wave = tid >> 5;
  const int half = lane >> 4;
  const int l16  = lane & 15;
  const int waveM = wave & 3;
  const int waveN = wave >> 2;
  const long mbase = (long)blockIdx.y * 128;
  const long nbase = (long)blockIdx.x * 128;

  // staging responsibilities
  const int arow = tid >> 1, ah = tid & 1;          // A: 32B chunk each
  const int bk = tid >> 3, bn0 = (tid & 7) * 16;    // B: 16 n-cols x 1 k each

  v8f acc[2][4];
#pragma unroll
  for (int mt = 0; mt < 2; mt++)
#pragma unroll
    for (int nt = 0; nt < 4; nt++) acc[mt][nt] = vzero8();

  // ---- prologue: stage buffer 0 for kb = 0 ----
  {
    const us* ga = A + (mbase + arow) * (long)K + ah * 16;
    us* da = &As[0][arow * 32 + ah * 16];
    async_ldg_b128(ga, da);
    async_ldg_b128(ga + 8, da + 8);
    U8 br0, br1;
    br0.q = *(const uint4*)(B + (long)bk * N + nbase + bn0);
    br1.q = *(const uint4*)(B + (long)bk * N + nbase + bn0 + 8);
#pragma unroll
    for (int e = 0; e < 8; e++) {
      Bs[0][(bn0 + e) * 32 + bk]     = br0.s[e];
      Bs[0][(bn0 + 8 + e) * 32 + bk] = br1.s[e];
    }
    wait_asynccnt0();
    __syncthreads();
  }

  for (int kb = 0; kb < K; kb += 32) {
    const int cur = (kb >> 5) & 1;
    const int nxt = cur ^ 1;
    const bool more = (kb + 32) < K;

    U8 br0, br1;
    if (more) {
      // async-DMA next A tile while this tile computes
      const us* ga = A + (mbase + arow) * (long)K + (kb + 32) + ah * 16;
      us* da = &As[nxt][arow * 32 + ah * 16];
      async_ldg_b128(ga, da);
      async_ldg_b128(ga + 8, da + 8);
      // next B tile into registers (transposed store later)
      br0.q = *(const uint4*)(B + (long)(kb + 32 + bk) * N + nbase + bn0);
      br1.q = *(const uint4*)(B + (long)(kb + 32 + bk) * N + nbase + bn0 + 8);
    }

    // ---- compute on current buffer: 2 Mt x 4 Nt WMMAs ----
    ABFrag af[2], bfr[4];
#pragma unroll
    for (int mt = 0; mt < 2; mt++) {
      const int row = waveM * 32 + mt * 16 + l16;
      af[mt].q[0] = *(const uint4*)(&As[cur][row * 32 + half * 8]);
      af[mt].q[1] = *(const uint4*)(&As[cur][row * 32 + 16 + half * 8]);
    }
#pragma unroll
    for (int nt = 0; nt < 4; nt++) {
      const int n = waveN * 64 + nt * 16 + l16;
      const uint4* p = (const uint4*)(&Bs[cur][n * 32 + half * 8]);
      bfr[nt].q[0] = p[0]; bfr[nt].q[1] = p[2];
    }
#pragma unroll
    for (int mt = 0; mt < 2; mt++)
#pragma unroll
      for (int nt = 0; nt < 4; nt++)
        acc[mt][nt] = wmma_bf16(af[mt].v, bfr[nt].v, acc[mt][nt]);

    if (more) {
#pragma unroll
      for (int e = 0; e < 8; e++) {
        Bs[nxt][(bn0 + e) * 32 + bk]     = br0.s[e];
        Bs[nxt][(bn0 + 8 + e) * 32 + bk] = br1.s[e];
      }
      wait_asynccnt0();
    }
    __syncthreads();
  }

  // epilogue: bias (+gelu) (+residual) -> f32 and/or bf16
#pragma unroll
  for (int mt = 0; mt < 2; mt++)
#pragma unroll
    for (int nt = 0; nt < 4; nt++) {
      const long col = nbase + waveN * 64 + nt * 16 + l16;
      const float bc = bias[col];
#pragma unroll
      for (int j = 0; j < 8; j++) {
        const long row = mbase + waveM * 32 + mt * 16 + j + half * 8;
        float vv = acc[mt][nt][j] + bc;
        if (gelu) vv = 0.5f * vv * (1.0f + erff(vv * 0.70710678118654752f));
        if (resid) vv += resid[row * N + col];
        if (outF) outF[row * N + col] = vv;
        if (outB) outB[row * N + col] = f2bf(vv);
      }
    }
}

// ---------------------------------------------------------------------------
// Flash attention: block = (qblock of 64 rows, head). 128 thr = 4 waves,
// each wave owns 16 query rows. Online softmax, WMMA for QK^T and PV.
// K tile staged via async-DMA; V tile register-transposed concurrently.
// ---------------------------------------------------------------------------
__global__ __launch_bounds__(128)
void attention_kernel(const us* __restrict__ Q, const us* __restrict__ Kg,
                      const us* __restrict__ Vg, us* __restrict__ O) {
  __shared__ us Kt[64 * 64];      // [key n][d]
  __shared__ us Vt[64 * 64];      // transposed: [d][key k]
  __shared__ us Pb[4][16 * 64];   // per-wave P tile [row][k]
  const int tid  = threadIdx.x;
  const int lane = tid & 31;
  const int wave = tid >> 5;
  const int half = lane >> 4;
  const int l16  = lane & 15;
  const int hh   = blockIdx.y;
  const long qrow0 = (long)blockIdx.x * 64 + wave * 16;

  // Q fragments for this wave's 16 rows, DH=64 -> 2 k-frags
  ABFrag qf[2];
  {
    const us* base = Q + (qrow0 + l16) * D_DIM + hh * DH;
#pragma unroll
    for (int f = 0; f < 2; f++) {
      qf[f].q[0] = *(const uint4*)(base + f * 32 + half * 8);
      qf[f].q[1] = *(const uint4*)(base + f * 32 + 16 + half * 8);
    }
  }

  v8f o[4];
#pragma unroll
  for (int dt = 0; dt < 4; dt++) o[dt] = vzero8();
  float mst[8], lst[8];
#pragma unroll
  for (int j = 0; j < 8; j++) { mst[j] = -3.0e38f; lst[j] = 0.0f; }

  for (int c = 0; c < 64; c++) {
    // stage K chunk via async DMA: Kt[n][d]
    {
      const int n = tid >> 1, h2 = tid & 1;
      const us* g = Kg + ((long)c * 64 + n) * D_DIM + hh * DH + h2 * 32;
      us* d = Kt + n * 64 + h2 * 32;
      async_ldg_b128(g, d);
      async_ldg_b128(g + 8, d + 8);
      async_ldg_b128(g + 16, d + 16);
      async_ldg_b128(g + 24, d + 24);
    }
    // stage V chunk transposed (VALU path, overlaps the K DMA): Vt[d][k]
    {
      const int r = tid >> 1, h2 = tid & 1;
      const uint4* g = (const uint4*)(Vg + ((long)c * 64 + r) * D_DIM + hh * DH + h2 * 32);
#pragma unroll
      for (int i = 0; i < 4; i++) {
        U8 u; u.q = g[i];
#pragma unroll
        for (int e = 0; e < 8; e++) Vt[(h2 * 32 + i * 8 + e) * 64 + r] = u.s[e];
      }
    }
    wait_asynccnt0();
    __syncthreads();

    // S = Q @ K^T   (16 x 64 per wave, 4 N-tiles x 2 K-steps)
    v8f sacc[4];
#pragma unroll
    for (int nt = 0; nt < 4; nt++) sacc[nt] = vzero8();
#pragma unroll
    for (int nt = 0; nt < 4; nt++)
#pragma unroll
      for (int f = 0; f < 2; f++) {
        ABFrag kf;
        const us* p = Kt + (nt * 16 + l16) * 64 + f * 32 + half * 8;
        kf.q[0] = ((const uint4*)p)[0];
        kf.q[1] = ((const uint4*)p)[2];
        sacc[nt] = wmma_bf16(qf[f].v, kf.v, sacc[nt]);
      }

    // online softmax (row reductions across each 16-lane half)
#pragma unroll
    for (int j = 0; j < 8; j++) {
      float cur = fmaxf(fmaxf(sacc[0][j], sacc[1][j]),
                        fmaxf(sacc[2][j], sacc[3][j])) * SCORE_SCALE;
#pragma unroll
      for (int m = 1; m < 16; m <<= 1) cur = fmaxf(cur, __shfl_xor(cur, m, 32));
      const float nm = fmaxf(mst[j], cur);
      const float alpha = __expf(mst[j] - nm);
      float rsum = 0.0f;
#pragma unroll
      for (int nt = 0; nt < 4; nt++) {
        const float p = __expf(sacc[nt][j] * SCORE_SCALE - nm);
        sacc[nt][j] = p;
        rsum += p;
      }
#pragma unroll
      for (int m = 1; m < 16; m <<= 1) rsum += __shfl_xor(rsum, m, 32);
      lst[j] = lst[j] * alpha + rsum;
      mst[j] = nm;
#pragma unroll
      for (int dt = 0; dt < 4; dt++) o[dt][j] *= alpha;
    }

    // re-layout P through per-wave LDS tile (C-layout -> A-fragment layout)
    us* pw = Pb[wave];
#pragma unroll
    for (int nt = 0; nt < 4; nt++)
#pragma unroll
      for (int j = 0; j < 8; j++)
        pw[(j + half * 8) * 64 + nt * 16 + l16] = f2bf(sacc[nt][j]);

    ABFrag pf[2];
#pragma unroll
    for (int f = 0; f < 2; f++) {
      const us* p = pw + l16 * 64 + f * 32 + half * 8;
      pf[f].q[0] = *(const uint4*)p;
      pf[f].q[1] = *(const uint4*)(p + 16);
    }

    // O += P @ V
#pragma unroll
    for (int dt = 0; dt < 4; dt++)
#pragma unroll
      for (int f = 0; f < 2; f++) {
        ABFrag vf;
        const us* p = Vt + (dt * 16 + l16) * 64 + f * 32 + half * 8;
        vf.q[0] = ((const uint4*)p)[0];
        vf.q[1] = ((const uint4*)p)[2];
        o[dt] = wmma_bf16(pf[f].v, vf.v, o[dt]);
      }
    __syncthreads();
  }

  // normalize and write attention output (bf16, (S,D) layout)
#pragma unroll
  for (int dt = 0; dt < 4; dt++)
#pragma unroll
    for (int j = 0; j < 8; j++) {
      const long s = qrow0 + j + half * 8;
      O[s * D_DIM + hh * DH + dt * 16 + l16] = f2bf(o[dt][j] * (1.0f / lst[j]));
    }
}

// ---------------------------------------------------------------------------
// Launch sequence
// ---------------------------------------------------------------------------
extern "C" void kernel_launch(void* const* d_in, const int* in_sizes, int n_in,
                              void* d_out, int out_size, void* d_ws, size_t ws_size,
                              hipStream_t stream) {
  (void)in_sizes; (void)n_in; (void)out_size; (void)ws_size;
  const float* x    = (const float*)d_in[0];
  const float* ln1g = (const float*)d_in[1];
  const float* ln1b = (const float*)d_in[2];
  const float* Wq   = (const float*)d_in[3];
  const float* bq   = (const float*)d_in[4];
  const float* Wk   = (const float*)d_in[5];
  const float* bk   = (const float*)d_in[6];
  const float* Wv   = (const float*)d_in[7];
  const float* bv   = (const float*)d_in[8];
  const float* Wo   = (const float*)d_in[9];
  const float* bo   = (const float*)d_in[10];
  const float* ln2g = (const float*)d_in[11];
  const float* ln2b = (const float*)d_in[12];
  const float* W1   = (const float*)d_in[13];
  const float* b1   = (const float*)d_in[14];
  const float* W2   = (const float*)d_in[15];
  const float* b2   = (const float*)d_in[16];
  float* out = (float*)d_out;

  char* ws = (char*)d_ws;
  const size_t MB = 1ull << 20;
  us*    WQb = (us*)(ws + 0 * MB);
  us*    WKb = (us*)(ws + 2 * MB);
  us*    WVb = (us*)(ws + 4 * MB);
  us*    WOb = (us*)(ws + 6 * MB);
  us*    W1b = (us*)(ws + 8 * MB);    // D*F bf16 = 8MB
  us*    W2b = (us*)(ws + 16 * MB);   // F*D bf16 = 8MB
  us*    XN  = (us*)(ws + 24 * MB);   // LN1(x) bf16
  us*    Qb  = (us*)(ws + 32 * MB);
  us*    Kb  = (us*)(ws + 40 * MB);
  us*    Vb  = (us*)(ws + 48 * MB);
  us*    ATb = (us*)(ws + 56 * MB);   // attention output bf16
  float* X2  = (float*)(ws + 64 * MB); // x + attn (fp32, 16MB)
  us*    HB  = (us*)(ws + 80 * MB);   // LN2 bf16
  us*    MID = (us*)(ws + 88 * MB);   // gelu(h@W1+b1) bf16, 32MB -> total 120MB

  // weights -> bf16
  f32_to_bf16_kernel<<<2048, 256, 0, stream>>>(Wq, WQb, (long)D_DIM * D_DIM);
  f32_to_bf16_kernel<<<2048, 256, 0, stream>>>(Wk, WKb, (long)D_DIM * D_DIM);
  f32_to_bf16_kernel<<<2048, 256, 0, stream>>>(Wv, WVb, (long)D_DIM * D_DIM);
  f32_to_bf16_kernel<<<2048, 256, 0, stream>>>(Wo, WOb, (long)D_DIM * D_DIM);
  f32_to_bf16_kernel<<<2048, 256, 0, stream>>>(W1, W1b, (long)D_DIM * F_DIM);
  f32_to_bf16_kernel<<<2048, 256, 0, stream>>>(W2, W2b, (long)F_DIM * D_DIM);

  // LN1
  layernorm_bf16_kernel<<<S_LEN, 256, 0, stream>>>(x, ln1g, ln1b, XN);

  // QKV projections
  dim3 gproj(D_DIM / 128, S_LEN / 128);   // (8, 32)
  gemm_bf16_kernel<<<gproj, 256, 0, stream>>>(XN, WQb, bq, nullptr, nullptr, Qb,
                                              S_LEN, D_DIM, D_DIM, 0);
  gemm_bf16_kernel<<<gproj, 256, 0, stream>>>(XN, WKb, bk, nullptr, nullptr, Kb,
                                              S_LEN, D_DIM, D_DIM, 0);
  gemm_bf16_kernel<<<gproj, 256, 0, stream>>>(XN, WVb, bv, nullptr, nullptr, Vb,
                                              S_LEN, D_DIM, D_DIM, 0);

  // attention
  dim3 gattn(S_LEN / 64, H_CNT);          // (64, 16)
  attention_kernel<<<gattn, 128, 0, stream>>>(Qb, Kb, Vb, ATb);

  // output projection + residual(x) -> X2 (fp32)
  gemm_bf16_kernel<<<gproj, 256, 0, stream>>>(ATb, WOb, bo, x, X2, nullptr,
                                              S_LEN, D_DIM, D_DIM, 0);

  // LN2
  layernorm_bf16_kernel<<<S_LEN, 256, 0, stream>>>(X2, ln2g, ln2b, HB);

  // FFN1 + exact GELU
  dim3 gffn1(F_DIM / 128, S_LEN / 128);   // (32, 32)
  gemm_bf16_kernel<<<gffn1, 256, 0, stream>>>(HB, W1b, b1, nullptr, nullptr, MID,
                                              S_LEN, F_DIM, D_DIM, 1);

  // FFN2 + residual(X2) -> out (fp32)
  gemm_bf16_kernel<<<gproj, 256, 0, stream>>>(MID, W2b, b2, X2, out, nullptr,
                                              S_LEN, D_DIM, F_DIM, 0);
}
